// GAT_60859686584469
// MI455X (gfx1250) — compile-verified, matrix operand order checked by reading, not run on hardware
//
#include <hip/hip_runtime.h>

// ---------------------------------------------------------------------------
// Types for CDNA5 WMMA
// ---------------------------------------------------------------------------
typedef __attribute__((ext_vector_type(16))) _Float16 v16h;
typedef __attribute__((ext_vector_type(8)))  _Float16 v8h;
typedef __attribute__((ext_vector_type(4)))  _Float16 v4h;
typedef __attribute__((ext_vector_type(8)))  float    v8f;

#define TILE_M 128
#define TILE_N 128
#define TILE_K 32
#define LDSTR  40   // halves per LDS row (32 + 8 pad) -> conflict-free b128 reads

// ---------------------------------------------------------------------------
// GEMM: C[M,Nc] = A[M,K] * B[K,Nc]. A is f16 row-major [M][K]; B is supplied
// pre-transposed f16 Bt[Nc][K]. f32 accumulate via v_wmma_f32_16x16x32_f16.
// 256 threads = 8 waves; each wave computes a 32x64 patch (2x4 fragments).
// LDS double-buffered; next tile's global b128 loads are issued before the
// current tile's WMMAs so load latency drains behind compute.
// ---------------------------------------------------------------------------
__global__ __launch_bounds__(256)
void gemm_f16wmma(const _Float16* __restrict__ A, const _Float16* __restrict__ Bt,
                  float* __restrict__ C, int M, int K, int Nc) {
  __shared__ __align__(16) _Float16 Ah[2][TILE_M * LDSTR];
  __shared__ __align__(16) _Float16 Bh[2][TILE_N * LDSTR];

  const int t       = threadIdx.x;
  const int rowBase = blockIdx.x * TILE_M;
  const int colBase = blockIdx.y * TILE_N;
  const bool fullA  = (rowBase + TILE_M) <= M;   // block-uniform guard

  // Tile-load chunk mapping: 2 chunks of 8 halves (16B) per thread per tile.
  const int r0 = t >> 2;            // 0..63
  const int c8 = (t & 3) * 8;       // 0,8,16,24

  const int wave = t >> 5, lane = t & 31;
  const int wm   = (wave >> 1) * 32;       // 0,32,64,96
  const int wn   = (wave & 1) * 64;        // 0,64
  const int lm   = lane & 15;
  const int kb   = (lane >> 4) << 3;       // 0 or 8 (ISA 16-bit A layout)

  v8f acc[2][4];
#pragma unroll
  for (int i = 0; i < 2; ++i)
#pragma unroll
    for (int j = 0; j < 4; ++j)
#pragma unroll
      for (int r = 0; r < 8; ++r) acc[i][j][r] = 0.f;

  v8h ra[2], rb[2];

  auto loadRegs = [&](int kk) {
#pragma unroll
    for (int i = 0; i < 2; ++i) {
      const int r  = r0 + i * 64;
      const int gr = rowBase + r;
      if (fullA || gr < M) {
        ra[i] = *(const v8h*)(A + (size_t)gr * K + kk + c8);
      } else {
#pragma unroll
        for (int q = 0; q < 8; ++q) ra[i][q] = (_Float16)0.f;
      }
      rb[i] = *(const v8h*)(Bt + (size_t)(colBase + r) * K + kk + c8);
    }
  };
  auto storeLDS = [&](int buf) {
#pragma unroll
    for (int i = 0; i < 2; ++i) {
      const int r = r0 + i * 64;
      *(v8h*)&Ah[buf][r * LDSTR + c8] = ra[i];
      *(v8h*)&Bh[buf][r * LDSTR + c8] = rb[i];
    }
  };

  loadRegs(0);
  storeLDS(0);
  __syncthreads();

  int cur = 0;
  for (int kk = 0; kk < K; kk += TILE_K) {
    const bool more = (kk + TILE_K) < K;
    if (more) loadRegs(kk + TILE_K);     // overlap with WMMAs below

    // Fragment loads: two contiguous b128 LDS reads per fragment.
    v16h afr[2], bfr[4];
#pragma unroll
    for (int i = 0; i < 2; ++i) {
      const _Float16* p = &Ah[cur][(wm + i * 16 + lm) * LDSTR];
      ((v8h*)&afr[i])[0] = *(const v8h*)(p + kb);        // K = kb..kb+7
      ((v8h*)&afr[i])[1] = *(const v8h*)(p + 16 + kb);   // K = 16+kb..16+kb+7
    }
#pragma unroll
    for (int j = 0; j < 4; ++j) {
      const _Float16* p = &Bh[cur][(wn + j * 16 + lm) * LDSTR];
      ((v8h*)&bfr[j])[0] = *(const v8h*)(p + kb);
      ((v8h*)&bfr[j])[1] = *(const v8h*)(p + 16 + kb);
    }
#pragma unroll
    for (int i = 0; i < 2; ++i)
#pragma unroll
      for (int j = 0; j < 4; ++j)
        acc[i][j] = __builtin_amdgcn_wmma_f32_16x16x32_f16(
            false, afr[i], false, bfr[j], (short)0, acc[i][j], false, false);

    if (more) {
      storeLDS(cur ^ 1);
      __syncthreads();
      cur ^= 1;
    }
  }

  // Epilogue: C/D layout (VGPR r: lanes0-15 -> M=r, lanes16-31 -> M=r+8).
  const int mOff = (lane >> 4) << 3;
  const int nOff = lane & 15;
  if (fullA) {
#pragma unroll
    for (int i = 0; i < 2; ++i)
#pragma unroll
      for (int j = 0; j < 4; ++j)
#pragma unroll
        for (int r = 0; r < 8; ++r) {
          const int gm = rowBase + wm + i * 16 + r + mOff;
          const int gn = colBase + wn + j * 16 + nOff;
          C[(size_t)gm * Nc + gn] = acc[i][j][r];
        }
  } else {
#pragma unroll
    for (int i = 0; i < 2; ++i)
#pragma unroll
      for (int j = 0; j < 4; ++j)
#pragma unroll
        for (int r = 0; r < 8; ++r) {
          const int gm = rowBase + wm + i * 16 + r + mOff;
          const int gn = colBase + wn + j * 16 + nOff;
          if (gm < M) C[(size_t)gm * Nc + gn] = acc[i][j][r];
        }
  }
}

// ---------------------------------------------------------------------------
// fp32 -> f16 streaming convert (float4 in, 4-half b64 out).
// ---------------------------------------------------------------------------
__global__ __launch_bounds__(256)
void cvt_f16_kernel(const float4* __restrict__ in, _Float16* __restrict__ out, int n4) {
  const int i = blockIdx.x * 256 + threadIdx.x;
  if (i >= n4) return;
  const float4 v = in[i];
  v4h o;
  o[0] = (_Float16)v.x; o[1] = (_Float16)v.y;
  o[2] = (_Float16)v.z; o[3] = (_Float16)v.w;
  *(v4h*)(out + (size_t)i * 4) = o;
}

// ---------------------------------------------------------------------------
// Wt[n][k] = (f16) W[k][n], 256x256.
// ---------------------------------------------------------------------------
__global__ __launch_bounds__(256)
void cvt_wT_kernel(const float* __restrict__ W, _Float16* __restrict__ Wt) {
  const int n = blockIdx.x, k = threadIdx.x;
  Wt[(size_t)n * 256 + k] = (_Float16)W[(size_t)k * 256 + n];
}

// ---------------------------------------------------------------------------
// Per-node attention dots: a_src[n,h] = sum_c xp[n,h*C+c]*att_src[h,c].
// ---------------------------------------------------------------------------
template <int C>
__global__ __launch_bounds__(256)
void attdot_kernel(const float* __restrict__ xp, const float* __restrict__ att_s,
                   const float* __restrict__ att_d, float* __restrict__ a_s,
                   float* __restrict__ a_d) {
  __shared__ float ss[256], sd[256];
  const int n = blockIdx.x, t = threadIdx.x;
  const float v = xp[(size_t)n * 256 + t];
  ss[t] = v * att_s[t];
  sd[t] = v * att_d[t];
  __syncthreads();
#pragma unroll
  for (int s = C >> 1; s > 0; s >>= 1) {
    if ((t & (C - 1)) < s) { ss[t] += ss[t + s]; sd[t] += sd[t + s]; }
    __syncthreads();
  }
  if ((t & (C - 1)) == 0) {
    constexpr int H = 256 / C;
    const int h = t / C;
    a_s[(size_t)n * H + h] = ss[t];
    a_d[(size_t)n * H + h] = sd[t];
  }
}

// ---------------------------------------------------------------------------
// Edge pass 1: e = leaky_relu(a_src[src]+a_dst[dst]); ordered-uint atomic max
// into m_enc[dst]. Edges >= E are self loops.
// ---------------------------------------------------------------------------
template <int H>
__global__ __launch_bounds__(256)
void edge_max_kernel(const int* __restrict__ ei, const float* __restrict__ asrc,
                     const float* __restrict__ adst, float* __restrict__ ebuf,
                     unsigned* __restrict__ menc, int E, int Etot) {
  const int tid = blockIdx.x * blockDim.x + threadIdx.x;
  if (tid >= Etot * H) return;
  const int edge = tid / H, h = tid - edge * H;
  int s, d;
  if (edge < E) { s = ei[edge]; d = ei[E + edge]; } else { s = d = edge - E; }
  float v = asrc[(size_t)s * H + h] + adst[(size_t)d * H + h];
  v = v > 0.f ? v : 0.2f * v;
  ebuf[tid] = v;                                    // tid == edge*H + h
  unsigned u = __float_as_uint(v);
  u = (u & 0x80000000u) ? ~u : (u | 0x80000000u);   // order-preserving encode
  atomicMax(&menc[(size_t)d * H + h], u);
}

// ---------------------------------------------------------------------------
// Edge pass 2: e = exp(e - m[dst]); atomicAdd denom[dst].
// ---------------------------------------------------------------------------
template <int H>
__global__ __launch_bounds__(256)
void edge_exp_kernel(const int* __restrict__ ei, float* __restrict__ ebuf,
                     const unsigned* __restrict__ menc, float* __restrict__ denom,
                     int E, int Etot) {
  const int tid = blockIdx.x * blockDim.x + threadIdx.x;
  if (tid >= Etot * H) return;
  const int edge = tid / H, h = tid - edge * H;
  const int d = (edge < E) ? ei[E + edge] : (edge - E);
  const unsigned u = menc[(size_t)d * H + h];
  const float m = (u & 0x80000000u) ? __uint_as_float(u & 0x7FFFFFFFu)
                                    : __uint_as_float(~u);
  const float ex = expf(ebuf[tid] - m);
  ebuf[tid] = ex;
  atomicAdd(&denom[(size_t)d * H + h], ex);
}

// ---------------------------------------------------------------------------
// Edge pass 3: agg[dst] += alpha * xp[src]. One wave per edge; 8 floats/lane.
// ---------------------------------------------------------------------------
template <int H, int C>
__global__ __launch_bounds__(256)
void edge_agg_kernel(const int* __restrict__ ei, const float* __restrict__ ebuf,
                     const float* __restrict__ denom, const float* __restrict__ xp,
                     float* __restrict__ agg, int E, int Etot) {
  const int edge = blockIdx.x * 8 + (threadIdx.x >> 5);
  if (edge >= Etot) return;
  const int lane = threadIdx.x & 31;
  int s, d;
  if (edge < E) { s = ei[edge]; d = ei[E + edge]; } else { s = d = edge - E; }
  const int f0 = lane * 8;
  const int h  = f0 / C;
  const float alpha = ebuf[(size_t)edge * H + h] /
                      (denom[(size_t)d * H + h] + 1e-16f);
  const float4* xs = (const float4*)(xp + (size_t)s * 256 + f0);
  const float4 v0 = xs[0], v1 = xs[1];
  float* o = agg + (size_t)d * 256 + f0;
  atomicAdd(o + 0, alpha * v0.x); atomicAdd(o + 1, alpha * v0.y);
  atomicAdd(o + 2, alpha * v0.z); atomicAdd(o + 3, alpha * v0.w);
  atomicAdd(o + 4, alpha * v1.x); atomicAdd(o + 5, alpha * v1.y);
  atomicAdd(o + 6, alpha * v1.z); atomicAdd(o + 7, alpha * v1.w);
}

// ---------------------------------------------------------------------------
// Layer-1 epilogue: h1h = f16( elu(agg + bias) ) -- f16 feeds the next GEMM.
// ---------------------------------------------------------------------------
__global__ __launch_bounds__(256)
void finish1_kernel(const float* __restrict__ agg, const float* __restrict__ bias,
                    _Float16* __restrict__ out) {
  const int t = threadIdx.x;
  const size_t i = (size_t)blockIdx.x * 256 + t;
  float v = agg[i] + bias[t];
  v = v > 0.f ? v : expm1f(v);
  out[i] = (_Float16)v;
}

// ---------------------------------------------------------------------------
// Final: ELU -> BatchNorm(eval) -> LayerNorm, one block per row, in place.
// ---------------------------------------------------------------------------
__global__ __launch_bounds__(256)
void finish2_kernel(float* __restrict__ out, const float* __restrict__ bias,
                    const float* __restrict__ bng, const float* __restrict__ bnb,
                    const float* __restrict__ bnm, const float* __restrict__ bnv,
                    const float* __restrict__ lng, const float* __restrict__ lnb) {
  __shared__ float red[256];
  const int t = threadIdx.x;
  const size_t i = (size_t)blockIdx.x * 256 + t;
  float v = out[i] + bias[t];
  v = v > 0.f ? v : expm1f(v);
  v = (v - bnm[t]) * rsqrtf(bnv[t] + 1e-5f) * bng[t] + bnb[t];
  red[t] = v;
  __syncthreads();
  for (int s = 128; s > 0; s >>= 1) { if (t < s) red[t] += red[t + s]; __syncthreads(); }
  const float mu = red[0] * (1.f / 256.f);
  __syncthreads();
  const float dv = v - mu;
  red[t] = dv * dv;
  __syncthreads();
  for (int s = 128; s > 0; s >>= 1) { if (t < s) red[t] += red[t + s]; __syncthreads(); }
  const float var = red[0] * (1.f / 256.f);
  out[i] = dv * rsqrtf(var + 1e-5f) * lng[t] + lnb[t];
}

// ---------------------------------------------------------------------------
// Launch
// ---------------------------------------------------------------------------
extern "C" void kernel_launch(void* const* d_in, const int* in_sizes, int n_in,
                              void* d_out, int out_size, void* d_ws, size_t ws_size,
                              hipStream_t stream) {
  const float* x   = (const float*)d_in[0];
  const int*   ei  = (const int*)  d_in[1];
  const float* W1  = (const float*)d_in[2];
  const float* as1 = (const float*)d_in[3];
  const float* ad1 = (const float*)d_in[4];
  const float* b1  = (const float*)d_in[5];
  const float* W2  = (const float*)d_in[6];
  const float* as2 = (const float*)d_in[7];
  const float* ad2 = (const float*)d_in[8];
  const float* b2  = (const float*)d_in[9];
  const float* bng = (const float*)d_in[10];
  const float* bnb = (const float*)d_in[11];
  const float* bnm = (const float*)d_in[12];
  const float* bnv = (const float*)d_in[13];
  const float* lng = (const float*)d_in[14];
  const float* lnb = (const float*)d_in[15];
  float* out = (float*)d_out;

  const int F    = 256;
  const int N    = in_sizes[0] / F;
  const int E    = in_sizes[1] / 2;
  const int Etot = E + N;

  // Workspace layout (float units; every segment is a multiple of 4 floats,
  // so all v8h/float4 accesses stay 16B aligned).
  float*    ws    = (float*)d_ws;
  const size_t NF = (size_t)N * F;
  float*    xp    = ws;                                   // N*256 f32
  float*    agg   = xp + NF;                              // N*256 f32
  float*    asrc  = agg + NF;                             // N*4
  float*    adst  = asrc + (size_t)N * 4;                 // N*4
  unsigned* menc  = (unsigned*)(adst + (size_t)N * 4);    // N*4
  float*    denom = (float*)(menc + (size_t)N * 4);       // N*4
  float*    ebuf  = denom + (size_t)N * 4;                // Etot*4
  _Float16* xh    = (_Float16*)(ebuf + (size_t)Etot * 4); // N*256 f16 (A matrix)
  _Float16* wt    = xh + NF;                              // 256*256 f16 (B^T)

  hipMemsetAsync(agg, 0, NF * sizeof(float), stream);
  hipMemsetAsync(out, 0, NF * sizeof(float), stream);

  const dim3 gblk(256);
  const dim3 ggrid((N + TILE_M - 1) / TILE_M, F / TILE_N);
  const int  n4 = (int)(NF / 4);

  // ---------------- Layer 1: H=4, C=64 ----------------
  cvt_f16_kernel<<<(n4 + 255) / 256, 256, 0, stream>>>((const float4*)x, xh, n4);
  cvt_wT_kernel<<<256, 256, 0, stream>>>(W1, wt);
  gemm_f16wmma<<<ggrid, gblk, 0, stream>>>(xh, wt, xp, N, F, F);
  attdot_kernel<64><<<N, 256, 0, stream>>>(xp, as1, ad1, asrc, adst);
  hipMemsetAsync(menc,  0, (size_t)N * 4 * sizeof(unsigned), stream);
  hipMemsetAsync(denom, 0, (size_t)N * 4 * sizeof(float), stream);
  {
    const int tot = Etot * 4;
    edge_max_kernel<4><<<(tot + 255) / 256, 256, 0, stream>>>(ei, asrc, adst, ebuf, menc, E, Etot);
    edge_exp_kernel<4><<<(tot + 255) / 256, 256, 0, stream>>>(ei, ebuf, menc, denom, E, Etot);
    edge_agg_kernel<4, 64><<<(Etot + 7) / 8, 256, 0, stream>>>(ei, ebuf, denom, xp, agg, E, Etot);
  }
  finish1_kernel<<<N, 256, 0, stream>>>(agg, b1, xh);   // xh now holds f16 h1

  // ---------------- Layer 2: H=1, C=256 ----------------
  cvt_wT_kernel<<<256, 256, 0, stream>>>(W2, wt);
  gemm_f16wmma<<<ggrid, gblk, 0, stream>>>(xh, wt, xp, N, F, F);
  attdot_kernel<256><<<N, 256, 0, stream>>>(xp, as2, ad2, asrc, adst);
  hipMemsetAsync(menc,  0, (size_t)N * sizeof(unsigned), stream);
  hipMemsetAsync(denom, 0, (size_t)N * sizeof(float), stream);
  {
    edge_max_kernel<1><<<(Etot + 255) / 256, 256, 0, stream>>>(ei, asrc, adst, ebuf, menc, E, Etot);
    edge_exp_kernel<1><<<(Etot + 255) / 256, 256, 0, stream>>>(ei, ebuf, menc, denom, E, Etot);
    edge_agg_kernel<1, 256><<<(Etot + 7) / 8, 256, 0, stream>>>(ei, ebuf, denom, xp, out, E, Etot);
  }
  finish2_kernel<<<N, 256, 0, stream>>>(out, b2, bng, bnb, bnm, bnv, lng, lnb);
}
